// DynamicFc_63625645523696
// MI455X (gfx1250) — matrix operand order repaired and checked by reading, not compiled, not run on hardware
//
#include <hip/hip_runtime.h>

// ---------------------------------------------------------------------------
// CDNA5 (gfx1250) fused DynamicFc. LOW=128, MID=32, F_DIM=1024, BATCH=8192.
// Never materialize the 256 MB `params`: regenerate dynamic weights per
// sample tile with bf16 WMMA against L2-resident pg_w (2 MB bf16), staged
// into LDS by the Tensor Data Mover (double-buffered, TENSORcnt-synced) and
// shared by 4 waves (64 samples) per workgroup.
// ---------------------------------------------------------------------------

typedef __attribute__((ext_vector_type(16))) __bf16 v16bf;
typedef __attribute__((ext_vector_type(8)))  float  v8f;
typedef __attribute__((ext_vector_type(4)))  unsigned u32x4;
typedef __attribute__((ext_vector_type(8)))  int      i32x8;
typedef __attribute__((ext_vector_type(4)))  int      i32x4;

#define BATCH 8192
#define FDIM  1024
#define LOW   128
#define MID   32
#define CH_ELEM 8192            // bf16 elements per staged chunk (16 KB)

#if defined(__HIP_DEVICE_COMPILE__) && \
    __has_builtin(__builtin_amdgcn_tensor_load_to_lds) && \
    __has_builtin(__builtin_amdgcn_s_wait_tensorcnt)
#define HAVE_TDM 1
#else
#define HAVE_TDM 0
#endif

// ISA 7.12.2: 16-bit A 16x32 — lane L holds row M=L&15; h=L>>4 selects the
// K subset. element j of v16bf -> K = (j&7) + 8*h + 16*(j>>3)  (interleaved)
__device__ __forceinline__ int kmapA(int j, int h) {
  return (j & 7) + 8 * h + ((j >> 3) << 4);
}
// 16-bit B 32x16 — lane holds col N=L&15; same K interleave as A
__device__ __forceinline__ int kmapB(int j, int h) { return (j & 7) + 8 * h + ((j >> 3) << 4); }

__device__ __forceinline__ v16bf loadA_f32(const float* __restrict__ p, int h) {
  v16bf v;
#pragma unroll
  for (int j = 0; j < 16; ++j) v[j] = (__bf16)p[kmapA(j, h)];
  return v;
}
__device__ __forceinline__ v16bf loadB_f32(const float* __restrict__ p, int h) {
  v16bf v;
#pragma unroll
  for (int j = 0; j < 16; ++j) v[j] = (__bf16)p[kmapB(j, h)];
  return v;
}
__device__ __forceinline__ v16bf loadA_bf16(const __bf16* p, int h) {
  v16bf v;
#pragma unroll
  for (int j = 0; j < 16; ++j) v[j] = p[kmapA(j, h)];
  return v;
}

__device__ __forceinline__ v8f wmma_bf16(v16bf a, v16bf b, v8f c) {
  return __builtin_amdgcn_wmma_f32_16x16x32_bf16(false, a, false, b,
                                                 (short)0, c, false, false);
}

// --------------------------- TDM staging -----------------------------------
#if HAVE_TDM
// 1-D contiguous tile, global -> LDS via Tensor Data Mover (ISA ch.8 D#).
__device__ __forceinline__ void tdm_load_1d(unsigned lds_off,
                                            const __bf16* gsrc, unsigned ne) {
  unsigned long long ga = (unsigned long long)gsrc;
  u32x4 g0;
  g0.x = 1u;                                            // count=1 descriptor
  g0.y = lds_off;                                       // lds_addr (bytes)
  g0.z = (unsigned)ga;                                  // global_addr[31:0]
  g0.w = (unsigned)((ga >> 32) & 0x01FFFFFFu) | (2u << 30);  // [56:32]|type=2
  i32x8 g1 = {};
  g1[0] = (1 << 16);                                    // data_size=1 (2 B)
  g1[1] = (int)((ne & 0xFFFFu) << 16);                  // tensor_dim0[15:0]
  g1[2] = (int)((ne >> 16) & 0xFFFFu) | (1 << 16);      // dim0[31:16]|dim1=1
  g1[3] = (int)((ne & 0xFFFFu) << 16);                  // tile_dim0
  g1[4] = 1;                                            // tile_dim1 = 1
  g1[5] = (int)ne;                                      // dim0_stride[31:0]
  i32x4 z4 = {};
#if __clang_major__ >= 23
  i32x8 z8 = {};
  __builtin_amdgcn_tensor_load_to_lds(g0, g1, z4, z4, z8, 0);
#else
  __builtin_amdgcn_tensor_load_to_lds(g0, g1, z4, z4, 0);
#endif
}
#endif

__device__ __forceinline__ void stage_chunk(__bf16* dst, const __bf16* src,
                                            int wave, int tid) {
#if HAVE_TDM
  if (wave == 0)
    tdm_load_1d((unsigned)(unsigned long long)(void*)dst, src, CH_ELEM);
#else
  const uint4* s = (const uint4*)src;
  uint4* d = (uint4*)dst;
  for (int i = tid; i < CH_ELEM * 2 / 16; i += 128) d[i] = s[i];
#endif
}
__device__ __forceinline__ void stage_fence() {
#if HAVE_TDM
  __builtin_amdgcn_s_wait_tensorcnt(0);
#endif
}

// ---------------------------------------------------------------------------
// Kernel 0: fp32 -> bf16 conversion of pg_w (once; halves on-chip traffic)
// ---------------------------------------------------------------------------
__global__ void k_cvt_bf16(const float* __restrict__ src,
                           __bf16* __restrict__ dst, int n) {
  int i = blockIdx.x * blockDim.x + threadIdx.x;
  if (i < n) dst[i] = (__bf16)src[i];
}

// ---------------------------------------------------------------------------
// Kernel 1: Y[b, n] = X[b, :] @ W[n, :] + bias[n]
// ---------------------------------------------------------------------------
__global__ __launch_bounds__(32) void k_proj(const float* __restrict__ X,
                                             const float* __restrict__ W,
                                             const float* __restrict__ bias,
                                             float* __restrict__ Y, int Kdim) {
  const int lane = threadIdx.x & 31;
  const int r = lane & 15, h = lane >> 4;
  const long row0 = (long)blockIdx.x * 16;
  const int n0 = blockIdx.y * 16;
  const float* arow = X + (row0 + r) * Kdim;
  const float* brow = W + (long)(n0 + r) * Kdim;
  v8f acc = {};
  for (int kk = 0; kk < Kdim; kk += 32) {
    acc = wmma_bf16(loadA_f32(arow + kk, h), loadB_f32(brow + kk, h), acc);
  }
  const int col = n0 + r;
  const float bv = bias[col];
#pragma unroll
  for (int i = 0; i < 8; ++i) {
    long row = row0 + i + 8 * h;
    Y[row * LOW + col] = acc[i] + bv;
  }
}

// ---------------------------------------------------------------------------
// Kernel 2: fused dynamic MLP. 4 waves x 16 samples = 64 samples / WG.
//  h[m,b] = relu( pg_b1^T@f_low^T + sum_l f_low[b,l]*(W1[l]@pf_low[b]) )
//  g[l,b] = pg_b2^T@relu(h)^T + sum_m relu(h[m,b])*(W2[m]@pf_low[b])
// pg_w chunks are TDM-DMA'd into double-buffered LDS, shared by all 4 waves.
// ---------------------------------------------------------------------------
__global__ __launch_bounds__(128) void k_dyn(const float* __restrict__ f_low,
                                             const float* __restrict__ pf_low,
                                             const __bf16* __restrict__ pgw,
                                             const float* __restrict__ pg_b,
                                             float* __restrict__ g_out) {
  __shared__ __bf16 s_stage[2 * CH_ELEM];   // 32 KB double buffer
  __shared__ float  s_h[4 * MID * 16];      // per-wave h transpose, 8 KB

  const int tid = threadIdx.x;
  const int lane = tid & 31;
  const int wave = tid >> 5;
  const int r = lane & 15, h = lane >> 4;
  const long b0 = (long)blockIdx.x * 64;
  const long bt = b0 + wave * 16;           // this wave's 16 samples
  float* mh = s_h + wave * (MID * 16);

  // B-operand fragments for this wave's samples: pf_low^T and f_low^T, K=128
  const float* prow = pf_low + (bt + r) * LOW;
  const float* frow = f_low + (bt + r) * LOW;
  v16bf Bf[4], Ff[4];
#pragma unroll
  for (int kc = 0; kc < 4; ++kc) {
    Bf[kc] = loadB_f32(prow + kc * 32, h);
    Ff[kc] = loadB_f32(frow + kc * 32, h);
  }

  // kick off first pg_w chunk DMA while we do the bias GEMM
  stage_chunk(s_stage, pgw, wave, tid);

  // ---- bias-1 GEMM: hacc = pg_b1^T (32x128) @ f_low^T (128x16) ------------
  float hacc[16];
#pragma unroll
  for (int mt = 0; mt < 2; ++mt) {
    v8f c = {};
#pragma unroll
    for (int kc = 0; kc < 4; ++kc) {
      v16bf a;
#pragma unroll
      for (int j = 0; j < 16; ++j)
        a[j] = (__bf16)pg_b[(kc * 32 + kmapA(j, h)) * MID + mt * 16 + r];
      c = wmma_bf16(a, Ff[kc], c);
    }
#pragma unroll
    for (int i = 0; i < 8; ++i) hacc[mt * 8 + i] = c[i];
  }

  // ---- phase 1: 64 chunks x 2 l-values, pure ds_load -> wmma --------------
  for (int c = 0; c < 64; ++c) {
    stage_fence();
    __syncthreads();
    if (c + 1 < 64)
      stage_chunk(s_stage + ((c + 1) & 1) * CH_ELEM,
                  pgw + (long)(c + 1) * CH_ELEM, wave, tid);
    const __bf16* sb = s_stage + (c & 1) * CH_ELEM;
#pragma unroll
    for (int li = 0; li < 2; ++li) {
      const int l = c * 2 + li;
      const float scale = f_low[(bt + r) * LOW + l];
#pragma unroll
      for (int mt = 0; mt < 2; ++mt) {
        v8f cc = {};
        const __bf16* wrow = sb + (li * MID + mt * 16 + r) * LOW;
#pragma unroll
        for (int kc = 0; kc < 4; ++kc)
          cc = wmma_bf16(loadA_bf16(wrow + kc * 32, h), Bf[kc], cc);
#pragma unroll
        for (int i = 0; i < 8; ++i) hacc[mt * 8 + i] += scale * cc[i];
      }
    }
  }
  __syncthreads();  // staging buffers free

  // prefetch first phase-2 chunk; overlap with h transpose + bias-2 GEMM
  const __bf16* pgw2 = pgw + (long)LOW * MID * LOW;
  stage_chunk(s_stage, pgw2, wave, tid);

  // h -> LDS (per-wave region) so relu(h) is readable per-lane and as B-frag
#pragma unroll
  for (int mt = 0; mt < 2; ++mt)
#pragma unroll
    for (int i = 0; i < 8; ++i)
      mh[(mt * 16 + i + 8 * h) * 16 + r] = hacc[mt * 8 + i];

  v16bf Bh;  // relu(h)^T as B operand, K = 32 m-values
#pragma unroll
  for (int j = 0; j < 16; ++j) {
    float v = mh[kmapB(j, h) * 16 + r];
    Bh[j] = (__bf16)(v > 0.f ? v : 0.f);
  }

  // ---- bias-2 GEMM: gacc = pg_b2^T (128x32) @ relu(h)^T (32x16) -----------
  float gacc[64];
#pragma unroll
  for (int lt = 0; lt < 8; ++lt) {
    v8f c = {};
    v16bf a;
#pragma unroll
    for (int j = 0; j < 16; ++j)
      a[j] = (__bf16)pg_b[LOW * MID + kmapA(j, h) * LOW + lt * 16 + r];
    c = wmma_bf16(a, Bh, c);
#pragma unroll
    for (int i = 0; i < 8; ++i) gacc[lt * 8 + i] = c[i];
  }

  // ---- phase 2: 64 chunks, chunk c = rows of m = c>>1, half = c&1 ---------
  for (int c = 0; c < 64; ++c) {
    stage_fence();
    __syncthreads();
    if (c + 1 < 64)
      stage_chunk(s_stage + ((c + 1) & 1) * CH_ELEM,
                  pgw2 + (long)(c + 1) * CH_ELEM, wave, tid);
    const __bf16* sb = s_stage + (c & 1) * CH_ELEM;
    const int m = c >> 1, half = c & 1;
    float hv = mh[m * 16 + r];
    const float s2 = hv > 0.f ? hv : 0.f;
#pragma unroll
    for (int t = 0; t < 4; ++t) {
      const int lt = half * 4 + t;
      v8f cc = {};
      const __bf16* wrow = sb + (t * 16 + r) * LOW;
#pragma unroll
      for (int kc = 0; kc < 4; ++kc)
        cc = wmma_bf16(loadA_bf16(wrow + kc * 32, h), Bf[kc], cc);
#pragma unroll
      for (int i = 0; i < 8; ++i) gacc[lt * 8 + i] += s2 * cc[i];
    }
  }
  __syncthreads();  // staging done; reuse s_stage as g transpose buffer

  // transpose g through LDS for coalesced b128 stores
  float* wg = (float*)s_stage + wave * (16 * LOW);
#pragma unroll
  for (int lt = 0; lt < 8; ++lt)
#pragma unroll
    for (int i = 0; i < 8; ++i)
      wg[r * LOW + lt * 16 + i + 8 * h] = gacc[lt * 8 + i];
  float4* sv = (float4*)wg;
  float4* dv = (float4*)(g_out + bt * LOW);
  for (int i = lane; i < 16 * LOW / 4; i += 32) dv[i] = sv[i];
}

// ---------------------------------------------------------------------------
// Kernel 3: out = g @ W2^T + b2 + f + pf   (M=BATCH, N=FDIM, K=128)
// ---------------------------------------------------------------------------
__global__ __launch_bounds__(32) void k_out(const float* __restrict__ g,
                                            const float* __restrict__ W2,
                                            const float* __restrict__ b2,
                                            const float* __restrict__ f,
                                            const float* __restrict__ pf,
                                            float* __restrict__ out) {
  const int lane = threadIdx.x & 31;
  const int r = lane & 15, h = lane >> 4;
  const long row0 = (long)blockIdx.x * 16;
  const int n0 = blockIdx.y * 16;
  const float* arow = g + (row0 + r) * LOW;
  const float* brow = W2 + (long)(n0 + r) * LOW;
  v8f acc = {};
#pragma unroll
  for (int kc = 0; kc < 4; ++kc)
    acc = wmma_bf16(loadA_f32(arow + kc * 32, h),
                    loadB_f32(brow + kc * 32, h), acc);
  const int col = n0 + r;
  const float bv = b2[col];
#pragma unroll
  for (int i = 0; i < 8; ++i) {
    long row = row0 + i + 8 * h;
    long idx = row * FDIM + col;
    out[idx] = acc[i] + bv + f[idx] + pf[idx];
  }
}

// ---------------------------------------------------------------------------
extern "C" void kernel_launch(void* const* d_in, const int* in_sizes, int n_in,
                              void* d_out, int out_size, void* d_ws,
                              size_t ws_size, hipStream_t stream) {
  const float* f         = (const float*)d_in[0];
  const float* pf        = (const float*)d_in[1];
  const float* proj_f_w  = (const float*)d_in[2];
  const float* proj_f_b  = (const float*)d_in[3];
  const float* proj_pf_w = (const float*)d_in[4];
  const float* proj_pf_b = (const float*)d_in[5];
  const float* proj_f2_w = (const float*)d_in[6];
  const float* proj_f2_b = (const float*)d_in[7];
  const float* pg_w      = (const float*)d_in[8];
  const float* pg_b      = (const float*)d_in[9];
  float* out = (float*)d_out;

  char* ws = (char*)d_ws;
  const size_t lowBytes = (size_t)BATCH * LOW * sizeof(float);  // 4 MB each
  float*  f_low  = (float*)(ws + 0 * lowBytes);
  float*  pf_low = (float*)(ws + 1 * lowBytes);
  float*  g_ws   = (float*)(ws + 2 * lowBytes);
  __bf16* pgw_bf = (__bf16*)(ws + 3 * lowBytes);                // 2 MB

  const int npg = 2 * LOW * MID * LOW;
  k_cvt_bf16<<<(npg + 255) / 256, 256, 0, stream>>>(pg_w, pgw_bf, npg);

  dim3 gp(BATCH / 16, LOW / 16);
  k_proj<<<gp, 32, 0, stream>>>(f,  proj_f_w,  proj_f_b,  f_low,  FDIM);
  k_proj<<<gp, 32, 0, stream>>>(pf, proj_pf_w, proj_pf_b, pf_low, FDIM);

  k_dyn<<<BATCH / 64, 128, 0, stream>>>(f_low, pf_low, pgw_bf, pg_b, g_ws);

  dim3 go(BATCH / 16, FDIM / 16);
  k_out<<<go, 32, 0, stream>>>(g_ws, proj_f2_w, proj_f2_b, f, pf, out);
}
